// SamplingNetwork_54219667145536
// MI455X (gfx1250) — matrix operand order repaired and checked by reading, not compiled, not run on hardware
//
#include <hip/hip_runtime.h>
#include <hip/hip_bf16.h>

#define B_    4
#define V_    32000
#define D_    256
#define NB_   4
#define FFNH  682
#define FFNP  704          // 682 padded to 22*32
#define ENCH  170
#define ENCP  192          // 170 padded to 6*32
#define TRH   85
#define TILE  128
#define NTILES ((B_ * V_) / TILE)   // 1000; 32000 % 128 == 0 so tiles never straddle batches

// padded LDS row strides (halfs; +8 keeps 16B alignment, breaks bank conflicts)
#define HTLD  (D_ + 8)      // 264
#define KPLD  (TILE + 8)    // 136
#define ACTLD (FFNP + 8)    // 712
#define XTLD  (2 * D_ + 8)  // 520
#define ATLD  (ENCP + 8)    // 200
#define NUMLD (D_ + 1)      // 257 (f32 scratch, scalar access)

typedef _Float16 f16;
typedef __attribute__((ext_vector_type(8)))  _Float16 v8h;
typedef __attribute__((ext_vector_type(16))) _Float16 v16h;
typedef __attribute__((ext_vector_type(8)))  float    v8f;

__device__ __forceinline__ v8f vzero() {
  v8f z;
#pragma unroll
  for (int i = 0; i < 8; ++i) z[i] = 0.f;
  return z;
}

__device__ __forceinline__ float siluf(float x)  { return x / (1.f + __expf(-x)); }
__device__ __forceinline__ float elu1f(float x)  { return x > 0.f ? x + 1.f : __expf(x); }
__device__ __forceinline__ float splusf(float x) { return x > 20.f ? x : log1pf(__expf(x)); }

// ---------------------------------------------------------------------------
// 16x16 tile accumulate over K (compile-time, multiple of 32) with
// v_wmma_f32_16x16x32_f16.
//   A: [M x K] row-major in LDS, row stride lda (halfs, multiple of 8)
//   B: [N x K] *N-major* in LDS, row stride ldb  -> per-lane fragment is
//      two contiguous 8-half runs (+0, +16) => 2x ds_load_b128 instead of 16x ds_load_u16.
// Per-lane layouts per CDNA5 ISA 7.12.2:
//   A: M=lane%16; halfs 0..7 -> K=8*hi+0..7, halfs 8..15 -> +16
//   B: N=lane%16; halfs 0..7 -> K=8*hi+0..7, halfs 8..15 -> +16
//   C: VGPR r -> M=r+8*hi, N=lane%16
// ---------------------------------------------------------------------------
template <int K>
__device__ __forceinline__ v8f wmma_tile(const f16* __restrict__ A, int lda, int m0,
                                         const f16* __restrict__ Bs, int ldb, int n0,
                                         v8f acc) {
  const int lane = threadIdx.x & 31;
  const int hi   = lane >> 4;
  const f16* arow = A + (m0 + (lane & 15)) * lda + hi * 8;
  const f16* brow = Bs + (n0 + (lane & 15)) * ldb + hi * 8;
#pragma unroll
  for (int k0 = 0; k0 < K; k0 += 32) {
    const v8h a0 = *(const v8h*)(arow + k0);
    const v8h a1 = *(const v8h*)(arow + k0 + 16);
    const v8h b0 = *(const v8h*)(brow + k0);
    const v8h b1 = *(const v8h*)(brow + k0 + 16);
    const v16h av = __builtin_shufflevector(a0, a1, 0, 1, 2, 3, 4, 5, 6, 7,
                                            8, 9, 10, 11, 12, 13, 14, 15);
    const v16h bv = __builtin_shufflevector(b0, b1, 0, 1, 2, 3, 4, 5, 6, 7,
                                            8, 9, 10, 11, 12, 13, 14, 15);
    acc = __builtin_amdgcn_wmma_f32_16x16x32_f16(false, av, false, bv,
                                                 (short)0, acc, false, false);
  }
  return acc;
}

// f32 global tile -> f16 LDS tile (row-major, padded stride), float4 ingest
__device__ __forceinline__ void load_tile_f16(const float* __restrict__ src,
                                              f16* __restrict__ dst, int ld) {
  for (int idx = threadIdx.x; idx < TILE * D_ / 4; idx += 256) {
    const int r = idx >> 6, c4 = (idx & 63) * 4;
    const float4 v = ((const float4*)src)[idx];
    f16* d = dst + r * ld + c4;
    d[0] = (f16)v.x; d[1] = (f16)v.y; d[2] = (f16)v.z; d[3] = (f16)v.w;
  }
}

// ---------------------------------------------------------------------------
// 1) log-softmax per batch row
// ---------------------------------------------------------------------------
__global__ void sn_logsoftmax(const float* __restrict__ logits, float* __restrict__ lp) {
  const int b = blockIdx.x;
  __shared__ float red[256];
  const float* x = logits + (size_t)b * V_;
  float m = -INFINITY;
  for (int v = threadIdx.x; v < V_; v += 256) m = fmaxf(m, x[v]);
  red[threadIdx.x] = m; __syncthreads();
  for (int s = 128; s > 0; s >>= 1) {
    if (threadIdx.x < s) red[threadIdx.x] = fmaxf(red[threadIdx.x], red[threadIdx.x + s]);
    __syncthreads();
  }
  m = red[0]; __syncthreads();
  float acc = 0.f;
  for (int v = threadIdx.x; v < V_; v += 256) acc += __expf(x[v] - m);
  red[threadIdx.x] = acc; __syncthreads();
  for (int s = 128; s > 0; s >>= 1) {
    if (threadIdx.x < s) red[threadIdx.x] += red[threadIdx.x + s];
    __syncthreads();
  }
  const float lse = m + logf(red[0]);
  for (int v = threadIdx.x; v < V_; v += 256) lp[(size_t)b * V_ + v] = x[v] - lse;
}

// ---------------------------------------------------------------------------
// 2) elementwise encoder SwiGLU (scalar -> ENCH -> D) via WMMA over 128 tokens
// ---------------------------------------------------------------------------
__global__ void sn_encoder(const float* __restrict__ lp,
                           const float* __restrict__ gw, const float* __restrict__ gb,
                           const float* __restrict__ uw, const float* __restrict__ ub,
                           const float* __restrict__ dw, const float* __restrict__ db,
                           float* __restrict__ hbuf) {
  extern __shared__ char sm[];
  f16* aT  = (f16*)sm;                       // [TILE][ATLD]
  f16* stg = aT + TILE * ATLD;               // [16][ATLD], N-major
  const int tok0 = blockIdx.x * TILE;
  const int wid  = threadIdx.x >> 5;
  const int lane = threadIdx.x & 31;

  for (int idx = threadIdx.x; idx < TILE * ENCP; idx += 256) {
    const int r = idx / ENCP, j = idx % ENCP;
    f16 val = (f16)0.f;
    if (j < ENCH) {
      const float x = lp[tok0 + r];
      val = (f16)(siluf(x * gw[j] + gb[j]) * (x * uw[j] + ub[j]));
    }
    aT[r * ATLD + j] = val;
  }
  __syncthreads();

  for (int ct = 0; ct < D_ / 16; ++ct) {
    const int c0 = ct * 16;
    for (int idx = threadIdx.x; idx < ENCP * 16; idx += 256) {
      const int k = idx >> 4, n = idx & 15;                 // coalesced read
      stg[n * ATLD + k] = (k < ENCH) ? (f16)dw[k * D_ + c0 + n] : (f16)0.f;
    }
    __syncthreads();
    v8f acc = wmma_tile<ENCP>(aT, ATLD, wid * 16, stg, ATLD, 0, vzero());
    const int col = c0 + (lane & 15);
    const float bias = db[col];
#pragma unroll
    for (int r = 0; r < 8; ++r) {
      const int row = wid * 16 + 8 * (lane >> 4) + r;
      hbuf[(size_t)(tok0 + row) * D_ + col] = acc[r] + bias;
    }
    __syncthreads();
  }
}

__global__ void sn_zero(float* __restrict__ p, int n) {
  const int i = blockIdx.x * 256 + threadIdx.x;
  if (i < n) p[i] = 0.f;
}

// ---------------------------------------------------------------------------
// 3a) pass A: QKV GEMM + elu+1; kp/v stored TRANSPOSED ([d][token]) so the
//     kv = kp^T @ v GEMM needs no strided access; kv/k_sum via fp32 atomics
// ---------------------------------------------------------------------------
__global__ void sn_qkv_kv(const float* __restrict__ hbuf,
                          const float* __restrict__ w, const float* __restrict__ bias,
                          f16* __restrict__ qp, float* __restrict__ kv,
                          float* __restrict__ ksum) {
  extern __shared__ char sm[];
  f16* ht  = (f16*)sm;                  // [TILE][HTLD]
  f16* kpT = ht + TILE * HTLD;          // [D][KPLD]  (kp transposed)
  f16* vvT = kpT + D_ * KPLD;           // [D][KPLD]  (v transposed)
  f16* stg = vvT + D_ * KPLD;           // [16][HTLD], N-major
  const int tok0 = blockIdx.x * TILE;
  const int b    = tok0 / V_;
  const int wid  = threadIdx.x >> 5;
  const int lane = threadIdx.x & 31;

  load_tile_f16(hbuf + (size_t)tok0 * D_, ht, HTLD);
  __syncthreads();

  for (int ct = 0; ct < (3 * D_) / 16; ++ct) {     // 48 col tiles over q|k|v
    const int c0 = ct * 16;
    if (lane == 0) __builtin_prefetch(&w[c0 + 16], 0, 3);
    for (int idx = threadIdx.x; idx < D_ * 16; idx += 256) {
      const int k = idx >> 4, n = idx & 15;
      stg[n * HTLD + k] = (f16)w[k * (3 * D_) + c0 + n];
    }
    __syncthreads();
    v8f acc = wmma_tile<D_>(ht, HTLD, wid * 16, stg, HTLD, 0, vzero());
    const int col = c0 + (lane & 15);
    const float bz = bias[col];
#pragma unroll
    for (int r = 0; r < 8; ++r) {
      const int row = wid * 16 + 8 * (lane >> 4) + r;
      const float val = acc[r] + bz;
      if (col < D_)          qp[(size_t)(tok0 + row) * D_ + col] = (f16)elu1f(val);
      else if (col < 2 * D_) kpT[(col - D_) * KPLD + row]        = (f16)elu1f(val);
      else                   vvT[(col - 2 * D_) * KPLD + row]    = (f16)val;
    }
    __syncthreads();
  }

  { // k_sum partial (contiguous reads thanks to transpose)
    const int d = threadIdx.x;
    float s = 0.f;
    for (int r = 0; r < TILE; ++r) s += (float)kpT[d * KPLD + r];
    unsafeAtomicAdd(&ksum[b * D_ + d], s);
  }

  // kv partial: [D x D] = kpT (M=d, K=token) @ vvT (N=e, K=token), K = 128
  for (int mt = wid; mt < D_ / 16; mt += 8) {
    for (int nt = 0; nt < D_ / 16; ++nt) {
      v8f acc = wmma_tile<TILE>(kpT, KPLD, mt * 16, vvT, KPLD, nt * 16, vzero());
      const int n = nt * 16 + (lane & 15);
#pragma unroll
      for (int r = 0; r < 8; ++r) {
        const int m = mt * 16 + 8 * (lane >> 4) + r;
        unsafeAtomicAdd(&kv[(size_t)b * D_ * D_ + m * D_ + n], acc[r]);
      }
    }
  }
}

// ---------------------------------------------------------------------------
// 3b) pass B1: num = qp @ kv (WMMA), den = qp . ksum, residual + LN1
// ---------------------------------------------------------------------------
__global__ void sn_attn_ln(float* __restrict__ hbuf, const f16* __restrict__ qp,
                           const float* __restrict__ kv, const float* __restrict__ ksum,
                           const float* __restrict__ g1, const float* __restrict__ b1) {
  extern __shared__ char sm[];
  f16*   qpl = (f16*)sm;                       // [TILE][HTLD]
  float* num = (float*)(qpl + TILE * HTLD);    // [TILE][NUMLD]
  f16*   stg = (f16*)(num + TILE * NUMLD);     // [16][HTLD], N-major
  const int tok0 = blockIdx.x * TILE;
  const int b    = tok0 / V_;
  const int wid  = threadIdx.x >> 5;
  const int lane = threadIdx.x & 31;

  // f16 tile ingest, 8 halfs (16B) per load
  for (int idx = threadIdx.x; idx < TILE * D_ / 8; idx += 256) {
    const int r = idx >> 5, c8 = (idx & 31) * 8;
    *(v8h*)(qpl + r * HTLD + c8) = ((const v8h*)(qp + (size_t)tok0 * D_))[idx];
  }
  __syncthreads();

  for (int ct = 0; ct < D_ / 16; ++ct) {
    const int c0 = ct * 16;
    for (int idx = threadIdx.x; idx < D_ * 16; idx += 256) {
      const int k = idx >> 4, n = idx & 15;
      stg[n * HTLD + k] = (f16)kv[(size_t)b * D_ * D_ + k * D_ + c0 + n];
    }
    __syncthreads();
    v8f acc = wmma_tile<D_>(qpl, HTLD, wid * 16, stg, HTLD, 0, vzero());
    const int col = c0 + (lane & 15);
#pragma unroll
    for (int r = 0; r < 8; ++r) {
      const int row = wid * 16 + 8 * (lane >> 4) + r;
      num[row * NUMLD + col] = acc[r];
    }
    __syncthreads();
  }

  if (threadIdx.x < TILE) {
    const int row = threadIdx.x;
    const size_t tok = tok0 + row;
    float den = 0.f;
    for (int d = 0; d < D_; ++d) den += (float)qpl[row * HTLD + d] * ksum[b * D_ + d];
    const float inv = 1.f / (den + 1e-6f);
    float mu = 0.f;
    for (int d = 0; d < D_; ++d) {
      const float x = hbuf[tok * D_ + d] + num[row * NUMLD + d] * inv;
      num[row * NUMLD + d] = x;
      mu += x;
    }
    mu *= (1.f / D_);
    float var = 0.f;
    for (int d = 0; d < D_; ++d) { const float dv = num[row * NUMLD + d] - mu; var += dv * dv; }
    const float rs = rsqrtf(var * (1.f / D_) + 1e-5f);
    for (int d = 0; d < D_; ++d)
      hbuf[tok * D_ + d] = (num[row * NUMLD + d] - mu) * rs * g1[d] + b1[d];
  }
}

// ---------------------------------------------------------------------------
// 3c) pass B2: SwiGLU FFN (256->682->256) + residual + LN2, all WMMA
// ---------------------------------------------------------------------------
__global__ void sn_ffn_ln(float* __restrict__ hbuf,
                          const float* __restrict__ gw, const float* __restrict__ gb,
                          const float* __restrict__ uw, const float* __restrict__ ub,
                          const float* __restrict__ dw, const float* __restrict__ db,
                          const float* __restrict__ g2, const float* __restrict__ b2) {
  extern __shared__ char sm[];
  f16* ht  = (f16*)sm;                  // [TILE][HTLD]
  f16* act = ht + TILE * HTLD;          // [TILE][ACTLD]
  f16* stg = act + TILE * ACTLD;        // max(32*HTLD, 16*ACTLD) = 11392 halfs
  const int tok0 = blockIdx.x * TILE;
  const int wid  = threadIdx.x >> 5;
  const int lane = threadIdx.x & 31;

  load_tile_f16(hbuf + (size_t)tok0 * D_, ht, HTLD);
  __syncthreads();

  // gate & up staged N-major as [32][HTLD] (rows 0-15 gate, 16-31 up)
  for (int ct = 0; ct < FFNP / 16; ++ct) {
    const int c0 = ct * 16;
    for (int idx = threadIdx.x; idx < D_ * 32; idx += 256) {
      const int k = idx >> 5, n = idx & 31;
      const int col = c0 + (n & 15);
      const float* src = (n < 16) ? gw : uw;
      stg[n * HTLD + k] = (col < FFNH) ? (f16)src[k * FFNH + col] : (f16)0.f;
    }
    __syncthreads();
    v8f accG = wmma_tile<D_>(ht, HTLD, wid * 16, stg, HTLD, 0,  vzero());
    v8f accU = wmma_tile<D_>(ht, HTLD, wid * 16, stg, HTLD, 16, vzero());
    const int col = c0 + (lane & 15);
    const float gbv = (col < FFNH) ? gb[col] : 0.f;
    const float ubv = (col < FFNH) ? ub[col] : 0.f;
#pragma unroll
    for (int r = 0; r < 8; ++r) {
      const int row = wid * 16 + 8 * (lane >> 4) + r;
      act[row * ACTLD + col] = (f16)(siluf(accG[r] + gbv) * (accU[r] + ubv));
    }
    __syncthreads();
  }

  // down projection [TILE x FFNP] @ [FFNP x D], zero-padded K rows
  for (int ct = 0; ct < D_ / 16; ++ct) {
    const int c0 = ct * 16;
    for (int idx = threadIdx.x; idx < FFNP * 16; idx += 256) {
      const int k = idx >> 4, n = idx & 15;
      stg[n * ACTLD + k] = (k < FFNH) ? (f16)dw[k * D_ + c0 + n] : (f16)0.f;
    }
    __syncthreads();
    v8f acc = wmma_tile<FFNP>(act, ACTLD, wid * 16, stg, ACTLD, 0, vzero());
    const int col = c0 + (lane & 15);
    const float bz = db[col];
#pragma unroll
    for (int r = 0; r < 8; ++r) {
      const int row = wid * 16 + 8 * (lane >> 4) + r;
      const float x = acc[r] + bz + (float)ht[row * HTLD + col];  // residual
      ht[row * HTLD + col] = (f16)x;                              // pre-LN, owned elem
    }
    __syncthreads();
  }

  if (threadIdx.x < TILE) {
    const int row = threadIdx.x;
    const size_t tok = tok0 + row;
    float mu = 0.f;
    for (int d = 0; d < D_; ++d) mu += (float)ht[row * HTLD + d];
    mu *= (1.f / D_);
    float var = 0.f;
    for (int d = 0; d < D_; ++d) { const float dv = (float)ht[row * HTLD + d] - mu; var += dv * dv; }
    const float rs = rsqrtf(var * (1.f / D_) + 1e-5f);
    for (int d = 0; d < D_; ++d)
      hbuf[tok * D_ + d] = ((float)ht[row * HTLD + d] - mu) * rs * g2[d] + b2[d];
  }
}

// ---------------------------------------------------------------------------
// 4) c = mean_v h  (coalesced: 256 threads = one d each, loop over v)
// ---------------------------------------------------------------------------
__global__ void sn_cmean(const float* __restrict__ hbuf, float* __restrict__ cmean) {
  const int b = blockIdx.x, d = threadIdx.x;
  float s = 0.f;
  for (int v = 0; v < V_; ++v) s += hbuf[((size_t)b * V_ + v) * D_ + d];
  cmean[b * D_ + d] = s * (1.f / V_);
}

// ---------------------------------------------------------------------------
// 5) truncation head (tiny, per batch): scale/threshold = softplus(swiglu(c))
// ---------------------------------------------------------------------------
__global__ void sn_trhead(const float* __restrict__ cmean,
                          const float* __restrict__ gw, const float* __restrict__ gb,
                          const float* __restrict__ uw, const float* __restrict__ ub,
                          const float* __restrict__ dw, const float* __restrict__ db,
                          float* __restrict__ trout) {
  const int b = blockIdx.x, j = threadIdx.x;
  __shared__ float a[TRH];
  if (j < TRH) {
    float g = gb[j], u = ub[j];
    for (int d = 0; d < D_; ++d) {
      const float c = cmean[b * D_ + d];
      g += c * gw[d * TRH + j];
      u += c * uw[d * TRH + j];
    }
    a[j] = siluf(g) * u;
  }
  __syncthreads();
  if (j < 2) {
    float s = db[j];
    for (int jj = 0; jj < TRH; ++jj) s += a[jj] * dw[jj * 2 + j];
    trout[b * 2 + j] = splusf(s);
  }
}

// ---------------------------------------------------------------------------
// 6) transformation head ([h|c] 512 -> 170 -> 1, WMMA) + final penalty/output
// ---------------------------------------------------------------------------
__global__ void sn_head(const float* __restrict__ hbuf, const float* __restrict__ cmean,
                        const float* __restrict__ lp,
                        const float* __restrict__ gw, const float* __restrict__ gb,
                        const float* __restrict__ uw, const float* __restrict__ ub,
                        const float* __restrict__ dw, const float* __restrict__ db,
                        const float* __restrict__ trout, float* __restrict__ out) {
  extern __shared__ char sm[];
  f16* xt  = (f16*)sm;                 // [TILE][XTLD]
  f16* act = xt + TILE * XTLD;         // [TILE][ATLD]
  f16* stg = act + TILE * ATLD;        // [32][XTLD], N-major
  const int tok0 = blockIdx.x * TILE;
  const int b    = tok0 / V_;
  const int wid  = threadIdx.x >> 5;
  const int lane = threadIdx.x & 31;

  for (int idx = threadIdx.x; idx < TILE * 2 * D_; idx += 256) {
    const int r = idx >> 9, c = idx & 511;
    xt[r * XTLD + c] = (f16)(c < D_ ? hbuf[(size_t)(tok0 + r) * D_ + c]
                                    : cmean[b * D_ + (c - D_)]);
  }
  __syncthreads();

  for (int ct = 0; ct < ENCP / 16; ++ct) {   // 12 col tiles over hidden
    const int c0 = ct * 16;
    for (int idx = threadIdx.x; idx < 2 * D_ * 32; idx += 256) {
      const int k = idx >> 5, n = idx & 31;
      const int col = c0 + (n & 15);
      const float* src = (n < 16) ? gw : uw;
      stg[n * XTLD + k] = (col < ENCH) ? (f16)src[k * ENCH + col] : (f16)0.f;
    }
    __syncthreads();
    v8f accG = wmma_tile<2 * D_>(xt, XTLD, wid * 16, stg, XTLD, 0,  vzero());
    v8f accU = wmma_tile<2 * D_>(xt, XTLD, wid * 16, stg, XTLD, 16, vzero());
    const int col = c0 + (lane & 15);
    const float gbv = (col < ENCH) ? gb[col] : 0.f;
    const float ubv = (col < ENCH) ? ub[col] : 0.f;
#pragma unroll
    for (int r = 0; r < 8; ++r) {
      const int row = wid * 16 + 8 * (lane >> 4) + r;
      act[row * ATLD + col] = (f16)(siluf(accG[r] + gbv) * (accU[r] + ubv));
    }
    __syncthreads();
  }

  if (threadIdx.x < TILE) {
    const int row = threadIdx.x;
    const size_t tok = tok0 + row;
    float lm = db[0];
    for (int j = 0; j < ENCH; ++j) lm += (float)act[row * ATLD + j] * dw[j];
    const float lprime = lp[tok] + lm;
    const float z = lprime * trout[b * 2 + 0] + trout[b * 2 + 1];
    out[tok] = lprime - splusf(-z);   // + log_sigmoid(z)
  }
}

// ---------------------------------------------------------------------------
extern "C" void kernel_launch(void* const* d_in, const int* in_sizes, int n_in,
                              void* d_out, int out_size, void* d_ws, size_t ws_size,
                              hipStream_t stream) {
  const float* logits = (const float*)d_in[0];
  const float* enc_gw = (const float*)d_in[1];
  const float* enc_gb = (const float*)d_in[2];
  const float* enc_uw = (const float*)d_in[3];
  const float* enc_ub = (const float*)d_in[4];
  const float* enc_dw = (const float*)d_in[5];
  const float* enc_db = (const float*)d_in[6];
  const float* qkv_w  = (const float*)d_in[7];
  const float* qkv_b  = (const float*)d_in[8];
  const float* ln1_g  = (const float*)d_in[9];
  const float* ln1_b  = (const float*)d_in[10];
  const float* ln2_g  = (const float*)d_in[11];
  const float* ln2_b  = (const float*)d_in[12];
  const float* ffn_gw = (const float*)d_in[13];
  const float* ffn_gb = (const float*)d_in[14];
  const float* ffn_uw = (const float*)d_in[15];
  const float* ffn_ub = (const float*)d_in[16];
  const float* ffn_dw = (const float*)d_in[17];
  const float* ffn_db = (const float*)d_in[18];
  const float* th_gw  = (const float*)d_in[19];
  const float* th_gb  = (const float*)d_in[20];
  const float* th_uw  = (const float*)d_in[21];
  const float* th_ub  = (const float*)d_in[22];
  const float* th_dw  = (const float*)d_in[23];
  const float* th_db  = (const float*)d_in[24];
  const float* tr_gw  = (const float*)d_in[25];
  const float* tr_gb  = (const float*)d_in[26];
  const float* tr_uw  = (const float*)d_in[27];
  const float* tr_ub  = (const float*)d_in[28];
  const float* tr_dw  = (const float*)d_in[29];
  const float* tr_db  = (const float*)d_in[30];

  // workspace layout
  float* lp    = (float*)d_ws;                           // B*V
  float* hbuf  = lp + (size_t)B_ * V_;                   // B*V*D f32
  f16*   qp    = (f16*)(hbuf + (size_t)B_ * V_ * D_);    // B*V*D f16
  float* kv    = (float*)(qp + (size_t)B_ * V_ * D_);    // B*D*D
  float* ksum  = kv + (size_t)B_ * D_ * D_;              // B*D
  float* cmean = ksum + B_ * D_;                         // B*D
  float* trout = cmean + B_ * D_;                        // B*2

  const size_t lds_enc = (size_t)(TILE * ATLD + 16 * ATLD) * sizeof(f16);
  const size_t lds_qkv = (size_t)(TILE * HTLD + 2 * D_ * KPLD + 16 * HTLD) * sizeof(f16);
  const size_t lds_att = (size_t)TILE * HTLD * sizeof(f16)
                       + (size_t)TILE * NUMLD * sizeof(float)
                       + (size_t)16 * HTLD * sizeof(f16);
  const size_t stg_ffn = (size_t)(32 * HTLD > 16 * ACTLD ? 32 * HTLD : 16 * ACTLD);
  const size_t lds_ffn = (size_t)(TILE * HTLD + TILE * ACTLD + stg_ffn) * sizeof(f16);
  const size_t lds_hd  = (size_t)(TILE * XTLD + TILE * ATLD + 32 * XTLD) * sizeof(f16);

  sn_logsoftmax<<<B_, 256, 0, stream>>>(logits, lp);
  sn_encoder<<<NTILES, 256, lds_enc, stream>>>(lp, enc_gw, enc_gb, enc_uw, enc_ub,
                                               enc_dw, enc_db, hbuf);
  for (int i = 0; i < NB_; ++i) {
    const int nz = B_ * D_ * D_ + B_ * D_;   // kv and ksum are contiguous
    sn_zero<<<(nz + 255) / 256, 256, 0, stream>>>(kv, nz);
    sn_qkv_kv<<<NTILES, 256, lds_qkv, stream>>>(hbuf,
        qkv_w + (size_t)i * D_ * 3 * D_, qkv_b + (size_t)i * 3 * D_, qp, kv, ksum);
    sn_attn_ln<<<NTILES, 256, lds_att, stream>>>(hbuf, qp, kv, ksum,
        ln1_g + (size_t)i * D_, ln1_b + (size_t)i * D_);
    sn_ffn_ln<<<NTILES, 256, lds_ffn, stream>>>(hbuf,
        ffn_gw + (size_t)i * D_ * FFNH, ffn_gb + (size_t)i * FFNH,
        ffn_uw + (size_t)i * D_ * FFNH, ffn_ub + (size_t)i * FFNH,
        ffn_dw + (size_t)i * FFNH * D_, ffn_db + (size_t)i * D_,
        ln2_g + (size_t)i * D_, ln2_b + (size_t)i * D_);
  }
  sn_cmean<<<B_, 256, 0, stream>>>(hbuf, cmean);
  sn_trhead<<<B_, 128, 0, stream>>>(cmean, tr_gw, tr_gb, tr_uw, tr_ub, tr_dw, tr_db, trout);
  sn_head<<<NTILES, 256, lds_hd, stream>>>(hbuf, cmean, lp, th_gw, th_gb, th_uw, th_ub,
                                           th_dw, th_db, trout, (float*)d_out);
}
